// Attention_60000693125929
// MI455X (gfx1250) — compile-verified, matrix operand order checked by reading, not run on hardware
//
#include <hip/hip_runtime.h>

// ---------------------------------------------------------------------------
// CDNA5 (gfx1250) attention block, WMMA f16 pipeline with fp32 accumulation.
//
//   0) convert w_qkv / w_out fp32 -> f16 (workspace)
//   1) RMSNorm(x) -> h (f16)
//   2) qkv = h @ w_qkv^T          (WMMA GEMM, async-LDS double-buffered)
//   3) RoPE on q,k (in place, f16)
//   4) flash attention            (WMMA QK^T and PV, online softmax)
//   5) out = o @ w_out^T          (WMMA GEMM, f32 out -> d_out)
//
// Fragment-layout convention (CDNA5 ISA, 16-bit A 16x32):
//   lane l: row = l&15, half = l>>4
//   v16h[0..7]  = K (8*half    .. 8*half+7)    -> one contiguous ds_load_b128
//   v16h[8..15] = K (16+8*half .. 16+8*half+7) -> one contiguous ds_load_b128
// B operand loaded with the same pattern applied to B^T rows (output columns
// supply K-major rows), matching y = x @ W^T with row-major W.
//
// Data movement: GLOBAL_LOAD_ASYNC_TO_LDS_B128 (ASYNCcnt) when the toolchain
// exposes the builtin; falls back to global_load_b128 + ds_store_b128.
// ---------------------------------------------------------------------------

typedef __attribute__((ext_vector_type(16))) _Float16 v16h;
typedef __attribute__((ext_vector_type(8)))  _Float16 v8h;
typedef __attribute__((ext_vector_type(8)))  float    v8f;
typedef __attribute__((ext_vector_type(4)))  int      v4i;

// address-space qualified int4 for the async-LDS builtin's pointer params
typedef __attribute__((address_space(1))) v4i gv4i;   // global
typedef __attribute__((address_space(3))) v4i lv4i;   // LDS

#if defined(__has_builtin)
#if __has_builtin(__builtin_amdgcn_global_load_async_to_lds_b128)
#define HAVE_ASYNC_LDS 1
#endif
#if __has_builtin(__builtin_amdgcn_s_wait_asynccnt)
#define HAVE_WAIT_ASYNC 1
#endif
#endif
#ifndef HAVE_ASYNC_LDS
#define HAVE_ASYNC_LDS 0
#endif
#ifndef HAVE_WAIT_ASYNC
#define HAVE_WAIT_ASYNC 0
#endif

#define DIM        2048
#define HEADS      16
#define DIM_HEAD   128
#define DIM_INNER  2048   // HEADS*DIM_HEAD
#define QKV_COLS   6144   // 3*DIM_INNER
#define SEQ        2048
#define BATCH      4
#define TOKENS     (BATCH*SEQ)   // 8192
#define RMS_EPS    1.1920929e-07f
#define SM_SCALE   0.08838834764831845f   // 1/sqrt(128)

// 16B global -> LDS copy; async (DMA, ASYNCcnt) when available.
__device__ __forceinline__ void copy16_g2l(const _Float16* g, _Float16* l) {
#if HAVE_ASYNC_LDS
  __builtin_amdgcn_global_load_async_to_lds_b128(
      (gv4i*)(void*)g, (lv4i*)(void*)l, 0, 0);
#else
  *(v8h*)l = *(const v8h*)g;
#endif
}

__device__ __forceinline__ void wait_g2l() {
#if HAVE_ASYNC_LDS
#if HAVE_WAIT_ASYNC
  __builtin_amdgcn_s_wait_asynccnt(0);
#else
  asm volatile("s_wait_asynccnt 0x0" ::: "memory");
#endif
#endif
}

__device__ __forceinline__ v16h make_frag(const _Float16* lo, const _Float16* hi) {
  v8h a = *(const v8h*)lo;
  v8h b = *(const v8h*)hi;
  v16h r;
#pragma unroll
  for (int i = 0; i < 8; ++i) { r[i] = a[i]; r[i + 8] = b[i]; }
  return r;
}

__device__ __forceinline__ v8f wmma_f16f32(v16h a, v16h b, v8f c) {
  return __builtin_amdgcn_wmma_f32_16x16x32_f16(false, a, false, b, (short)0, c,
                                                false, false);
}

// ---------------------------------------------------------------------------
// 0) fp32 -> f16 convert (weights)
// ---------------------------------------------------------------------------
__global__ __launch_bounds__(256) void cvt_f32_f16(const float* __restrict__ a,
                                                   _Float16* __restrict__ o,
                                                   int n) {
  int i = blockIdx.x * 256 + threadIdx.x;
  if (i < n) o[i] = (_Float16)a[i];
}

// ---------------------------------------------------------------------------
// 1) RMSNorm: one 256-thread block per token row (2048 cols, 8 per thread)
// ---------------------------------------------------------------------------
__global__ __launch_bounds__(256) void rmsnorm_f16(const float* __restrict__ x,
                                                   const float* __restrict__ w,
                                                   _Float16* __restrict__ h) {
  __shared__ float red[8];
  const int row = blockIdx.x;
  const float* xr = x + (size_t)row * DIM;
  float v[8], ss = 0.f;
#pragma unroll
  for (int i = 0; i < 8; ++i) { v[i] = xr[threadIdx.x + i * 256]; ss += v[i] * v[i]; }
#pragma unroll
  for (int m = 16; m >= 1; m >>= 1) ss += __shfl_xor(ss, m, 32);
  if ((threadIdx.x & 31) == 0) red[threadIdx.x >> 5] = ss;
  __syncthreads();
  float tot = 0.f;
#pragma unroll
  for (int i = 0; i < 8; ++i) tot += red[i];
  const float scale = rsqrtf(tot * (1.f / DIM) + RMS_EPS);
  _Float16* hr = h + (size_t)row * DIM;
#pragma unroll
  for (int i = 0; i < 8; ++i) {
    int c = threadIdx.x + i * 256;
    hr[c] = (_Float16)(v[i] * scale * w[c]);
  }
}

// ---------------------------------------------------------------------------
// 2/5) GEMM  C[M,N] = A[M,K] * B[N,K]^T   (A,B f16 row-major, K%32==0)
//      128x128 tile / block (256 thr = 8 waves), 16-row strip per wave.
//      Double-buffered LDS staging via async global->LDS DMA.
// ---------------------------------------------------------------------------
template <bool OUT_F32>
__global__ __launch_bounds__(256) void gemm_xwT(const _Float16* __restrict__ A,
                                                const _Float16* __restrict__ B,
                                                void* __restrict__ Cv,
                                                int M, int N, int K) {
  __shared__ _Float16 As[2][128][40];   // 32 K + 8 pad, x2 buffers
  __shared__ _Float16 Bs[2][128][40];
  const int n0 = blockIdx.x * 128;
  const int m0 = blockIdx.y * 128;
  const int tid = threadIdx.x;
  const int wave = tid >> 5, lane = tid & 31;
  const int half = lane >> 4, l16 = lane & 15;

  auto stage = [&](int buf, int k0) {
#pragma unroll
    for (int c = 0; c < 2; ++c) {              // 512 8-half chunks per operand
      int chunk = tid + c * 256;
      int r = chunk >> 2;
      int kk = (chunk & 3) * 8;
      copy16_g2l(&A[(size_t)(m0 + r) * K + k0 + kk], &As[buf][r][kk]);
      copy16_g2l(&B[(size_t)(n0 + r) * K + k0 + kk], &Bs[buf][r][kk]);
    }
  };

  v8f acc[8];
#pragma unroll
  for (int i = 0; i < 8; ++i)
#pragma unroll
    for (int j = 0; j < 8; ++j) acc[i][j] = 0.f;

  stage(0, 0);
  wait_g2l();
  __syncthreads();

  int cur = 0;
  for (int k0 = 0; k0 < K; k0 += 32) {
    const bool more = (k0 + 32) < K;
    if (more) stage(cur ^ 1, k0 + 32);     // DMA next tile while computing
    const _Float16* ap = &As[cur][16 * wave + l16][8 * half];
    v16h afrag = make_frag(ap, ap + 16);
#pragma unroll
    for (int nn = 0; nn < 8; ++nn) {
      const _Float16* bp = &Bs[cur][nn * 16 + l16][8 * half];
      v16h bfrag = make_frag(bp, bp + 16);
      acc[nn] = wmma_f16f32(afrag, bfrag, acc[nn]);
    }
    if (more) wait_g2l();
    __syncthreads();
    cur ^= 1;
  }

#pragma unroll
  for (int nn = 0; nn < 8; ++nn)
#pragma unroll
    for (int r = 0; r < 8; ++r) {
      int m = m0 + 16 * wave + r + 8 * half;
      int n = n0 + nn * 16 + l16;
      if constexpr (OUT_F32)
        ((float*)Cv)[(size_t)m * N + n] = acc[nn][r];
      else
        ((_Float16*)Cv)[(size_t)m * N + n] = (_Float16)acc[nn][r];
    }
}

// ---------------------------------------------------------------------------
// 3) RoPE on q,k (in place).  One thread per (qk, b, n, head, pair).
// ---------------------------------------------------------------------------
__global__ __launch_bounds__(256) void rope_f16(_Float16* __restrict__ qkv,
                                                const float* __restrict__ freqs) {
  long p = (long)blockIdx.x * 256 + threadIdx.x;  // 16,777,216 total
  int d2 = (int)(p & 63);  long t = p >> 6;
  int hh = (int)(t & 15);  t >>= 4;
  int n  = (int)(t & 2047); t >>= 11;
  int b  = (int)(t & 3);
  int qk = (int)(t >> 2);
  size_t off = ((size_t)(b * SEQ + n)) * QKV_COLS + (size_t)qk * DIM_INNER +
               hh * DIM_HEAD + 2 * d2;
  float f = freqs[n * DIM_HEAD + 2 * d2];
  float s, c;
  __sincosf(f, &s, &c);
  float x0 = (float)qkv[off], x1 = (float)qkv[off + 1];
  qkv[off]     = (_Float16)(x0 * c - x1 * s);
  qkv[off + 1] = (_Float16)(x1 * c + x0 * s);
}

// ---------------------------------------------------------------------------
// 4) Flash attention. Block = (b, head, 128 query rows), 8 waves x 16 rows.
//    Key loop in 32-wide tiles; S via 8 WMMA, PV via 8 WMMA per tile.
//    K tile staged via async global->LDS; V transposed through VGPR scatter.
// ---------------------------------------------------------------------------
__global__ __launch_bounds__(256) void flash_attn(const _Float16* __restrict__ qkv,
                                                  _Float16* __restrict__ O) {
  __shared__ _Float16 Qs[128][136];       // 34,816 B
  __shared__ _Float16 Ks[32][136];        //  8,704 B
  __shared__ _Float16 Vt[128][40];        // 10,240 B (V transposed: [d][j])
  __shared__ _Float16 Ps[8][16][40];      // 10,240 B (per-wave P bounce)

  const int m0 = blockIdx.x * 128;
  const int hh = blockIdx.y;
  const int b  = blockIdx.z;
  const int tid = threadIdx.x;
  const int wave = tid >> 5, lane = tid & 31;
  const int half = lane >> 4, l16 = lane & 15;

  const _Float16* Qg = qkv + (size_t)b * SEQ * QKV_COLS + hh * DIM_HEAD;
  const _Float16* Kg = Qg + DIM_INNER;
  const _Float16* Vg = Qg + 2 * DIM_INNER;

  // stage Q block (128 x 128 halves) via async DMA
#pragma unroll
  for (int c = 0; c < 8; ++c) {
    int chunk = tid + c * 256;            // 0..2047
    int r = chunk >> 4;
    int dd = (chunk & 15) * 8;
    copy16_g2l(&Qg[(size_t)(m0 + r) * QKV_COLS + dd], &Qs[r][dd]);
  }
  wait_g2l();
  __syncthreads();

  const int mrow = 16 * wave;
  v16h qfrag[4];
#pragma unroll
  for (int kc = 0; kc < 4; ++kc) {
    const _Float16* p = &Qs[mrow + l16][kc * 32 + 8 * half];
    qfrag[kc] = make_frag(p, p + 16);
  }

  v8f acc[8];
  float mx_[8], lsum_[8];
#pragma unroll
  for (int i = 0; i < 8; ++i) {
    mx_[i] = -3.0e38f; lsum_[i] = 0.f;
#pragma unroll
    for (int j = 0; j < 8; ++j) acc[i][j] = 0.f;
  }

  const int jtiles = (blockIdx.x + 1) * 4;   // causal: keys up to m0+127
  for (int jt = 0; jt < jtiles; ++jt) {
    const int j0 = jt * 32;
    // cooperative K/V tile staging (32 x 128 each)
#pragma unroll
    for (int c = 0; c < 2; ++c) {
      int chunk = tid + c * 256;
      int r = chunk >> 4;
      int dd = (chunk & 15) * 8;
      copy16_g2l(&Kg[(size_t)(j0 + r) * QKV_COLS + dd], &Ks[r][dd]);
      v8h vv = *(const v8h*)&Vg[(size_t)(j0 + r) * QKV_COLS + dd];
#pragma unroll
      for (int i = 0; i < 8; ++i) Vt[dd + i][r] = vv[i];   // transpose into LDS
    }
    wait_g2l();
    __syncthreads();

    if (j0 <= m0 + mrow + 15) {            // wave-level causal early-out
      v8f s0, s1;
#pragma unroll
      for (int i = 0; i < 8; ++i) { s0[i] = 0.f; s1[i] = 0.f; }
#pragma unroll
      for (int kc = 0; kc < 4; ++kc) {
        const _Float16* b0 = &Ks[l16][kc * 32 + 8 * half];
        const _Float16* b1 = &Ks[16 + l16][kc * 32 + 8 * half];
        s0 = wmma_f16f32(qfrag[kc], make_frag(b0, b0 + 16), s0);
        s1 = wmma_f16f32(qfrag[kc], make_frag(b1, b1 + 16), s1);
      }
      const int col0 = j0 + l16, col1 = j0 + 16 + l16;
#pragma unroll
      for (int r = 0; r < 8; ++r) {
        const int rowg = m0 + mrow + r + 8 * half;
        const bool ok0 = (col0 <= rowg), ok1 = (col1 <= rowg);
        float a0 = ok0 ? s0[r] * SM_SCALE : -3.0e38f;
        float a1 = ok1 ? s1[r] * SM_SCALE : -3.0e38f;
        float mt = fmaxf(a0, a1);
#pragma unroll
        for (int msk = 8; msk >= 1; msk >>= 1) mt = fmaxf(mt, __shfl_xor(mt, msk, 32));
        const float mnew = fmaxf(mx_[r], mt);
        const float alpha = __expf(mx_[r] - mnew);
        mx_[r] = mnew;
        const float p0 = ok0 ? __expf(a0 - mnew) : 0.f;   // explicit zero on mask
        const float p1 = ok1 ? __expf(a1 - mnew) : 0.f;
        s0[r] = p0; s1[r] = p1;
        float ps = p0 + p1;
#pragma unroll
        for (int msk = 8; msk >= 1; msk >>= 1) ps += __shfl_xor(ps, msk, 32);
        lsum_[r] = lsum_[r] * alpha + ps;
#pragma unroll
        for (int dd = 0; dd < 8; ++dd) acc[dd][r] *= alpha;
      }
      // C-layout fp32 P -> A-layout f16 via wave-private LDS patch
#pragma unroll
      for (int r = 0; r < 8; ++r) {
        Ps[wave][r + 8 * half][l16]      = (_Float16)s0[r];
        Ps[wave][r + 8 * half][16 + l16] = (_Float16)s1[r];
      }
      const _Float16* pp = &Ps[wave][l16][8 * half];
      v16h pfrag = make_frag(pp, pp + 16);
#pragma unroll
      for (int dd = 0; dd < 8; ++dd) {
        const _Float16* vp = &Vt[dd * 16 + l16][8 * half];
        acc[dd] = wmma_f16f32(pfrag, make_frag(vp, vp + 16), acc[dd]);
      }
    }
    __syncthreads();
  }

  // write O in [b, n, h*128+d] layout (feeds out-proj GEMM directly)
  _Float16* Og = O + (size_t)b * SEQ * DIM_INNER + hh * DIM_HEAD;
#pragma unroll
  for (int r = 0; r < 8; ++r) {
    const float inv = 1.f / lsum_[r];     // diagonal guarantees lsum > 0
    const int rowg = m0 + mrow + r + 8 * half;
#pragma unroll
    for (int dd = 0; dd < 8; ++dd)
      Og[(size_t)rowg * DIM_INNER + dd * 16 + l16] = (_Float16)(acc[dd][r] * inv);
  }
}

// ---------------------------------------------------------------------------
// launcher
// ---------------------------------------------------------------------------
extern "C" void kernel_launch(void* const* d_in, const int* in_sizes, int n_in,
                              void* d_out, int out_size, void* d_ws, size_t ws_size,
                              hipStream_t stream) {
  const float* x      = (const float*)d_in[0];   // [4,2048,2048]
  const float* freqs  = (const float*)d_in[1];   // [2048,128]
  const float* w_rms  = (const float*)d_in[2];   // [2048]
  const float* w_qkv  = (const float*)d_in[3];   // [6144,2048]
  const float* w_out  = (const float*)d_in[4];   // [2048,2048]
  float* out = (float*)d_out;

  // f16 staging in workspace (~192 MiB total)
  char* ws = (char*)d_ws;
  _Float16* h_f16    = (_Float16*)(ws);                         // 33.5 MB
  _Float16* wqkv_f16 = (_Float16*)(ws + (size_t)33554432);      // 25.2 MB
  _Float16* qkv_f16  = (_Float16*)(ws + (size_t)58720256);      // 100.7 MB
  _Float16* wout_f16 = (_Float16*)(ws + (size_t)159383552);     // 8.4 MB
  _Float16* o_f16    = (_Float16*)(ws + (size_t)167772160);     // 33.5 MB

  // 0) weight converts
  {
    int n1 = QKV_COLS * DIM;
    cvt_f32_f16<<<(n1 + 255) / 256, 256, 0, stream>>>(w_qkv, wqkv_f16, n1);
    int n2 = DIM * DIM_INNER;
    cvt_f32_f16<<<(n2 + 255) / 256, 256, 0, stream>>>(w_out, wout_f16, n2);
  }
  // 1) RMSNorm
  rmsnorm_f16<<<TOKENS, 256, 0, stream>>>(x, w_rms, h_f16);
  // 2) QKV GEMM: [8192,2048] x [6144,2048]^T -> [8192,6144] f16
  gemm_xwT<false><<<dim3(QKV_COLS / 128, TOKENS / 128), 256, 0, stream>>>(
      h_f16, wqkv_f16, (void*)qkv_f16, TOKENS, QKV_COLS, DIM);
  // 3) RoPE on q,k
  {
    long pairs = 2L * BATCH * SEQ * HEADS * (DIM_HEAD / 2);   // 16,777,216
    rope_f16<<<(unsigned)(pairs / 256), 256, 0, stream>>>(qkv_f16, freqs);
  }
  // 4) flash attention
  flash_attn<<<dim3(SEQ / 128, HEADS, BATCH), 256, 0, stream>>>(qkv_f16, o_f16);
  // 5) out projection: [8192,2048] x [2048,2048]^T -> fp32 d_out
  gemm_xwT<true><<<dim3(DIM / 128, TOKENS / 128), 256, 0, stream>>>(
      o_f16, wout_f16, (void*)out, TOKENS, DIM, DIM_INNER);
}
